// MessageUpdate_38130719654482
// MI455X (gfx1250) — matrix operand order repaired and checked by reading, not compiled, not run on hardware
//
#include <hip/hip_runtime.h>
#include <hip/hip_bf16.h>

typedef __attribute__((ext_vector_type(16))) _Float16 v16h;
typedef __attribute__((ext_vector_type(8)))  _Float16 v8h;
typedef __attribute__((ext_vector_type(4)))  _Float16 v4h;
typedef __attribute__((ext_vector_type(8)))  float    v8f;

#define DIN   128
#define DB    64
#define DO    128
#define T_TYPES 4
#define DCAT  320            // 2*DIN + DB
#define MT    8              // M-tiles (of 16 edges) per block
#define ROWS  (MT * 16)      // 128 edges per block
#define WM    2              // M-tiles per wave
#define WN    2              // N-tiles per wave
#define THREADS 512          // 16 waves: 4 M-groups x 4 N-groups
#define PER_TYPE_HALVES 114688   // (320*128)*2 + (128*128)*2
#define OFF_W1A 0
#define OFF_W1B 40960
#define OFF_W2A 57344
#define OFF_W2B 98304

#define DEV_INLINE __device__ __attribute__((always_inline)) inline

// ---- fragment loaders (CDNA5 WMMA 16x16x32 f16 layouts, ISA 7.12.2) ----

// 16 contiguous halves (pre-swizzled B fragment) at base + 32-bit offset;
// base stays a uniform (scalar) pointer -> global_load_b128 saddr form.
DEV_INLINE v16h ld_frag16(const _Float16* __restrict__ base, int off) {
    const _Float16* p = base + off;
    v8h lo = *(const v8h*)p;
    v8h hi = *(const v8h*)(p + 8);
    v16h r;
#pragma unroll
    for (int j = 0; j < 8; ++j) { r[j] = lo[j]; r[j + 8] = hi[j]; }
    return r;
}

// A fragment from LDS, row-major [16][stride] f16:
// lane<16:  M=lane,    halves = K kb+0..7  and kb+16..23
// lane>=16: M=lane-16, halves = K kb+8..15 and kb+24..31
DEV_INLINE v16h ld_afrag(const _Float16* s, int stride, int kb, int lane) {
    int row = lane & 15;
    int klo = (lane & 16) ? 8 : 0;
    const _Float16* p = s + row * stride + kb + klo;
    v8h lo = *(const v8h*)p;
    v8h hi = *(const v8h*)(p + 16);
    v16h r;
#pragma unroll
    for (int j = 0; j < 8; ++j) { r[j] = lo[j]; r[j + 8] = hi[j]; }
    return r;
}

// One 2-layer MLP pass for a 128-edge tile. Wave `wave` owns a 2Mx2N grid of
// 16x16 tiles: M-tiles (wave>>2)*2+i, N-tiles (wave&3)*2+j. A and B fragments
// are each reused across the register tile.
DEV_INLINE void mlp_pass(const _Float16* __restrict__ sV,
                         _Float16* sH, float* sO,
                         const _Float16* __restrict__ wA,
                         const _Float16* __restrict__ wB,
                         const float* __restrict__ ba,
                         const float* __restrict__ bb,
                         int wave, int lane) {
    const int mg    = wave >> 2;         // 0..3
    const int ng    = wave & 3;          // 0..3
    const int lcol  = lane & 15;
    const int rbase = (lane & 16) ? 8 : 0;

    // layer 1: [128,320] x [320,128]
    v8f acc[WM][WN];
#pragma unroll
    for (int i = 0; i < WM; ++i)
#pragma unroll
        for (int j = 0; j < WN; ++j) acc[i][j] = (v8f){};
#pragma unroll
    for (int ks = 0; ks < DCAT / 32; ++ks) {
        v16h bfr[WN];
#pragma unroll
        for (int j = 0; j < WN; ++j) {
            int boff = (((ng * WN + j) * (DCAT / 32) + ks) * 32 + lane) * 16;
            bfr[j] = ld_frag16(wA, boff);
        }
        v16h afr[WM];
#pragma unroll
        for (int i = 0; i < WM; ++i)
            afr[i] = ld_afrag(sV + (mg * WM + i) * 16 * DCAT, DCAT, ks * 32, lane);
#pragma unroll
        for (int i = 0; i < WM; ++i)
#pragma unroll
            for (int j = 0; j < WN; ++j)
                acc[i][j] = __builtin_amdgcn_wmma_f32_16x16x32_f16(
                    false, afr[i], false, bfr[j], (short)0, acc[i][j], false, false);
    }
#pragma unroll
    for (int j = 0; j < WN; ++j) {
        int col = (ng * WN + j) * 16 + lcol;
        float bias = ba[col];
#pragma unroll
        for (int i = 0; i < WM; ++i)
#pragma unroll
            for (int r = 0; r < 8; ++r) {
                float x = acc[i][j][r] + bias;
                x = (x > 0.f) ? x : 0.01f * x;       // leaky_relu
                sH[((mg * WM + i) * 16 + rbase + r) * DO + col] = (_Float16)x;
            }
    }
    __syncthreads();

    // layer 2: [128,128] x [128,128]
    v8f acc2[WM][WN];
#pragma unroll
    for (int i = 0; i < WM; ++i)
#pragma unroll
        for (int j = 0; j < WN; ++j) acc2[i][j] = (v8f){};
#pragma unroll
    for (int ks = 0; ks < DO / 32; ++ks) {
        v16h bfr[WN];
#pragma unroll
        for (int j = 0; j < WN; ++j) {
            int boff = (((ng * WN + j) * (DO / 32) + ks) * 32 + lane) * 16;
            bfr[j] = ld_frag16(wB, boff);
        }
        v16h afr[WM];
#pragma unroll
        for (int i = 0; i < WM; ++i)
            afr[i] = ld_afrag(sH + (mg * WM + i) * 16 * DO, DO, ks * 32, lane);
#pragma unroll
        for (int i = 0; i < WM; ++i)
#pragma unroll
            for (int j = 0; j < WN; ++j)
                acc2[i][j] = __builtin_amdgcn_wmma_f32_16x16x32_f16(
                    false, afr[i], false, bfr[j], (short)0, acc2[i][j], false, false);
    }
#pragma unroll
    for (int j = 0; j < WN; ++j) {
        int col = (ng * WN + j) * 16 + lcol;
        float bias = bb[col];
#pragma unroll
        for (int i = 0; i < WM; ++i)
#pragma unroll
            for (int r = 0; r < 8; ++r) {
                float x = acc2[i][j][r] + bias;
                x = (x > 0.f) ? x : 0.01f * x;       // leaky_relu (outer)
                sO[((mg * WM + i) * 16 + rbase + r) * DO + col] = x;
            }
    }
    __syncthreads();
}

// ---------------- prep kernels ----------------

__global__ void k_init(int* meta) {
    if (threadIdx.x < 12) meta[threadIdx.x] = 0;     // cnt[4], off[4], cursor[4]
}

__global__ void k_zero(float* p, int n) {
    int i = blockIdx.x * 256 + threadIdx.x;
    if (i < n) p[i] = 0.f;
}

__global__ void k_count(const int* __restrict__ uc, int* meta, int E) {
    int e = blockIdx.x * 256 + threadIdx.x;
    if (e < E) atomicAdd(&meta[uc[e]], 1);
}

__global__ void k_scan(int* meta) {
    if (threadIdx.x == 0) {
        int s = 0;
        for (int i = 0; i < T_TYPES; ++i) { meta[4 + i] = s; s += meta[i]; }
    }
}

__global__ void k_bucket(const int* __restrict__ uc, int* meta, int* eord, int E) {
    int e = blockIdx.x * 256 + threadIdx.x;
    if (e < E) {
        int t = uc[e];
        int pos = atomicAdd(&meta[8 + t], 1);
        eord[meta[4 + t] + pos] = e;
    }
}

// f32 [T][K][DO] -> pre-swizzled f16 B fragments.
// Linear dst index = ((nt*ksteps + ks)*32 + lane)*16 + j  maps to
// K = ks*32 + (lane<16 ? 0 : 8) + (j&7) + (j>=8 ? 16 : 0),  N = nt*16 + (lane&15).
__global__ void k_swz(const float* __restrict__ src, _Float16* dst,
                      int K, int perTypeDstOff) {
    int t   = blockIdx.y;
    int lin = blockIdx.x * 256 + threadIdx.x;        // < K*DO
    int ksteps = K / 32;
    int j    = lin & 15;
    int lane = (lin >> 4) & 31;
    int ks   = (lin >> 9) % ksteps;
    int nt   = lin / (512 * ksteps);
    int n    = nt * 16 + (lane & 15);
    int k    = ks * 32 + ((lane & 16) ? 8 : 0) + (j & 7) + ((j & 8) ? 16 : 0);
    dst[(size_t)t * PER_TYPE_HALVES + perTypeDstOff + lin] =
        (_Float16)src[((size_t)t * K + k) * DO + n];
}

// ---------------- main fused kernel ----------------
// grid.x = T * tilesPerType (128-edge tiles), grid.y = B, block = 512 (16 waves)
__global__ void __launch_bounds__(THREADS)
k_edge_mlp(const float* __restrict__ sites, const float* __restrict__ bonds,
           const int* __restrict__ idx1, const int* __restrict__ idx2,
           const float* __restrict__ b1a, const float* __restrict__ b1b,
           const float* __restrict__ b2a, const float* __restrict__ b2b,
           const float* __restrict__ Wa1, const float* __restrict__ ba1,
           const float* __restrict__ Wa2, const float* __restrict__ ba2,
           const int* __restrict__ meta, const int* __restrict__ eord,
           const _Float16* __restrict__ wsw,
           float* agg, int N, int E, int tilesPerType) {
    __shared__ __align__(16) _Float16 sV[ROWS * DCAT];   // 80 KB
    __shared__ __align__(16) _Float16 sH[ROWS * DO];     // 32 KB
    __shared__ __align__(16) float    sO1[ROWS * DO];    // 64 KB
    __shared__ __align__(16) float    sO2[ROWS * DO];    // 64 KB
    __shared__ int   sE[ROWS];
    __shared__ int   sN1[ROWS];
    __shared__ int   sN2[ROWS];
    __shared__ float sDot[2 * ROWS];

    const int t    = blockIdx.x / tilesPerType;
    const int tile = blockIdx.x % tilesPerType;
    const int b    = blockIdx.y;
    const int cnt  = meta[t];
    if (tile * ROWS >= cnt) return;                  // block-uniform early exit
    const int off  = meta[4 + t];
    const int tid  = threadIdx.x;
    const int lane = tid & 31;
    const int wave = tid >> 5;

    if (tid < ROWS) {
        int r = tile * ROWS + tid;
        if (r < cnt) {
            int e = eord[off + r];
            sE[tid]  = e;
            sN1[tid] = idx1[e];
            sN2[tid] = idx2[e];
        } else {
            sE[tid] = -1; sN1[tid] = 0; sN2[tid] = 0;
        }
    }
    __syncthreads();

    // gather v = [sites[idx1] | sites[idx2] | bonds] -> f16 LDS (float4 loads)
    for (int idx = tid; idx < ROWS * (DCAT / 4); idx += THREADS) {
        int row = idx / (DCAT / 4);
        int c   = (idx - row * (DCAT / 4)) * 4;
        int e   = sE[row];
        float4 val = make_float4(0.f, 0.f, 0.f, 0.f);
        if (e >= 0) {
            if (c < DIN)
                val = *(const float4*)(sites + ((size_t)b * N + sN1[row]) * DIN + c);
            else if (c < 2 * DIN)
                val = *(const float4*)(sites + ((size_t)b * N + sN2[row]) * DIN + (c - DIN));
            else
                val = *(const float4*)(bonds + ((size_t)b * E + e) * DB + (c - 2 * DIN));
        }
        v4h h;
        h[0] = (_Float16)val.x; h[1] = (_Float16)val.y;
        h[2] = (_Float16)val.z; h[3] = (_Float16)val.w;
        *(v4h*)(sV + row * DCAT + c) = h;
    }
    __syncthreads();

    const _Float16* wt = wsw + (size_t)t * PER_TYPE_HALVES;
    mlp_pass(sV, sH, sO1, wt + OFF_W1A, wt + OFF_W1B, b1a + t * DO, b1b + t * DO, wave, lane);
    mlp_pass(sV, sH, sO2, wt + OFF_W2A, wt + OFF_W2B, b2a + t * DO, b2b + t * DO, wave, lane);

    // attention dots: threads 0..127 -> out1 rows, 128..255 -> out2 rows
    if (tid < 2 * ROWS) {
        int r = tid & (ROWS - 1);
        const float* src = (tid < ROWS) ? sO1 : sO2;
        const float* wv  = (tid < ROWS) ? Wa1 : Wa2;
        float s = 0.f;
        for (int c = 0; c < DO; c += 4) {
            float4 v = *(const float4*)(src + r * DO + c);
            float4 w = *(const float4*)(wv + c);
            s += v.x * w.x + v.y * w.y + v.z * w.z + v.w * w.w;
        }
        sDot[tid] = s;
    }
    __syncthreads();

    // sigmoid gate + scatter-add into agg[:, idx2]
#pragma unroll
    for (int m = 0; m < MT / 2; ++m) {               // 4 x 32 rows
        int row = m * 32 + (tid >> 4);
        int c0  = (tid & 15) * 8;
        int e   = sE[row];
        if (e >= 0) {
            float g1 = 1.f / (1.f + __expf(-(sDot[row]        + ba1[0])));
            float g2 = 1.f / (1.f + __expf(-(sDot[ROWS + row] + ba2[0])));
            float* dst = agg + ((size_t)b * N + sN2[row]) * DO;
#pragma unroll
            for (int j = 0; j < 8; ++j) {
                int c = c0 + j;
                float g = g1 * sO1[row * DO + c] + g2 * sO2[row * DO + c];
                atomicAdd(dst + c, g);
            }
        }
    }
}

// ---------------- host launch ----------------

extern "C" void kernel_launch(void* const* d_in, const int* in_sizes, int n_in,
                              void* d_out, int out_size, void* d_ws, size_t ws_size,
                              hipStream_t stream) {
    const float* sites = (const float*)d_in[0];
    const float* bonds = (const float*)d_in[1];
    const int*   idx1  = (const int*)d_in[2];
    const int*   idx2  = (const int*)d_in[3];
    const int*   uc    = (const int*)d_in[4];
    const float* W1a   = (const float*)d_in[5];
    const float* b1a   = (const float*)d_in[6];
    const float* W1b   = (const float*)d_in[7];
    const float* b1b   = (const float*)d_in[8];
    const float* W2a   = (const float*)d_in[9];
    const float* b2a   = (const float*)d_in[10];
    const float* W2b   = (const float*)d_in[11];
    const float* b2b   = (const float*)d_in[12];
    const float* Wa1   = (const float*)d_in[13];
    const float* ba1   = (const float*)d_in[14];
    const float* Wa2   = (const float*)d_in[15];
    const float* ba2   = (const float*)d_in[16];

    const int E = in_sizes[2];
    const int B = in_sizes[1] / (E * DB);
    const int N = in_sizes[0] / (B * DIN);

    // ws layout: int cnt[4] | off[4] | cursor[4] | pad | eord[E] | f16 weights
    int* meta = (int*)d_ws;
    int* eord = meta + 16;
    size_t wsoff = ((64 + (size_t)E * 4 + 255) / 256) * 256;
    _Float16* wsw = (_Float16*)((char*)d_ws + wsoff);
    float* agg = (float*)d_out;

    k_init<<<1, 32, 0, stream>>>(meta);
    k_zero<<<(out_size + 255) / 256, 256, 0, stream>>>(agg, out_size);
    k_count<<<(E + 255) / 256, 256, 0, stream>>>(uc, meta, E);
    k_scan<<<1, 1, 0, stream>>>(meta);
    k_bucket<<<(E + 255) / 256, 256, 0, stream>>>(uc, meta, eord, E);

    dim3 gBig(DCAT * DO / 256, T_TYPES);
    dim3 gSml(DO * DO / 256, T_TYPES);
    k_swz<<<gBig, 256, 0, stream>>>(W1a, wsw, DCAT, OFF_W1A);
    k_swz<<<gSml, 256, 0, stream>>>(W1b, wsw, DO,   OFF_W1B);
    k_swz<<<gBig, 256, 0, stream>>>(W2a, wsw, DCAT, OFF_W2A);
    k_swz<<<gSml, 256, 0, stream>>>(W2b, wsw, DO,   OFF_W2B);

    int tilesPerType = (E + ROWS - 1) / ROWS;
    dim3 grid(T_TYPES * tilesPerType, B);
    k_edge_mlp<<<grid, THREADS, 0, stream>>>(sites, bonds, idx1, idx2,
                                             b1a, b1b, b2a, b2b,
                                             Wa1, ba1, Wa2, ba2,
                                             meta, eord, wsw, agg,
                                             N, E, tilesPerType);
}
